// SAGE_34333968564343
// MI455X (gfx1250) — matrix operand order repaired and checked by reading, not compiled, not run on hardware
//
#include <hip/hip_runtime.h>

#define N_NODES 50000
#define N_EDGES 600000
#define IN_F 128          // both layers have K = 128
#define HID_F 128
#define OUT_F 64

typedef __attribute__((ext_vector_type(16))) _Float16 v16h;
typedef __attribute__((ext_vector_type(8)))  float    v8f;

// ---------------------------------------------------------------- utilities
__global__ __launch_bounds__(256) void sage_zero_kernel(float* __restrict__ p, int n) {
    int i = blockIdx.x * blockDim.x + threadIdx.x;
    if (i < n) p[i] = 0.0f;
}

__global__ __launch_bounds__(256) void sage_deg_kernel(const int* __restrict__ dst,
                                                       float* __restrict__ deg) {
    int e = blockIdx.x * blockDim.x + threadIdx.x;
    if (e < N_EDGES) atomicAdd(&deg[dst[e]], 1.0f);
}

// Convert fp32 weight [128 x out_feats] (row-major, K x N) to f16, pre-swizzled
// into this file's WMMA B-fragment order (contiguous k, same as A; not the ISA table order):
//   fragment(tile_n, tile_k): lane holds column N = tile_n*16 + (lane&15),
//   element i holds K = tile_k*32 + (lane>>4)*16 + i.
// Linear layout: (((tile_n*4 + tile_k)*32 + lane)*16 + i)
__global__ __launch_bounds__(256) void sage_cvtw_kernel(const float* __restrict__ W,
                                                        _Float16* __restrict__ Wf,
                                                        int out_feats) {
    int idx = blockIdx.x * blockDim.x + threadIdx.x;
    int total = IN_F * out_feats;
    if (idx >= total) return;
    int i      = idx & 15;
    int lane   = (idx >> 4) & 31;
    int t      = idx >> 9;          // 512 halves per fragment
    int tile_k = t & 3;             // K = 128 -> 4 K-tiles of 32
    int tile_n = t >> 2;
    int k = tile_k * 32 + ((lane >> 4) << 4) + i;
    int n = tile_n * 16 + (lane & 15);
    Wf[idx] = (_Float16)W[k * out_feats + n];
}

// wave (32 lanes) per edge; lane l moves feats [4l, 4l+4)
__global__ __launch_bounds__(256) void sage_scatter_kernel(const float* __restrict__ h,
                                                           const int* __restrict__ src,
                                                           const int* __restrict__ dst,
                                                           float* __restrict__ agg) {
    long gid = (long)blockIdx.x * blockDim.x + threadIdx.x;
    int e = (int)(gid >> 5);
    if (e >= N_EDGES) return;
    int c = (int)(gid & 31) << 2;
    const float4 v = *(const float4*)(h + (long)src[e] * IN_F + c);
    float* o = agg + (long)dst[e] * IN_F + c;
    atomicAdd(o + 0, v.x);
    atomicAdd(o + 1, v.y);
    atomicAdd(o + 2, v.z);
    atomicAdd(o + 3, v.w);
}

// ------------------------------------------------- fused normalize + WMMA GEMM
// Block = 256 threads = 8 waves. Block handles 16 rows (M-tile). Wave w owns
// N-tile [16w, 16w+16). A tile (agg+h)/(deg+1) staged in LDS as f16.
__global__ __launch_bounds__(256) void sage_gemm_kernel(const float* __restrict__ agg,
                                                        const float* __restrict__ h,
                                                        const float* __restrict__ deg,
                                                        const _Float16* __restrict__ Wf,
                                                        const float* __restrict__ bias,
                                                        float* __restrict__ out,
                                                        int out_feats, int do_relu) {
    __shared__ _Float16 As[16 * IN_F];           // 4 KB
    const int tid = threadIdx.x;
    const int m0  = blockIdx.x * 16;             // 50000 = 3125*16, no remainder

    // --- stage + normalize A tile: 2048 halves, 8 per thread (same row each)
    {
        const int idx = tid << 3;                // 0..2040
        const int row = idx >> 7;                // 0..15
        const long g  = (long)(m0 + row) * IN_F + (idx & 127);
        const float r = 1.0f / (deg[m0 + row] + 1.0f);
        const float4 a0 = *(const float4*)(agg + g);
        const float4 x0 = *(const float4*)(h + g);
        const float4 a1 = *(const float4*)(agg + g + 4);
        const float4 x1 = *(const float4*)(h + g + 4);
        _Float16* s = As + idx;
        s[0] = (_Float16)((a0.x + x0.x) * r);
        s[1] = (_Float16)((a0.y + x0.y) * r);
        s[2] = (_Float16)((a0.z + x0.z) * r);
        s[3] = (_Float16)((a0.w + x0.w) * r);
        s[4] = (_Float16)((a1.x + x1.x) * r);
        s[5] = (_Float16)((a1.y + x1.y) * r);
        s[6] = (_Float16)((a1.z + x1.z) * r);
        s[7] = (_Float16)((a1.w + x1.w) * r);
    }
    __syncthreads();

    const int wave = tid >> 5;
    const int lane = tid & 31;
    const int n0   = wave << 4;
    if (n0 < out_feats) {                        // wave-uniform: EXEC all-1s inside
        const int half = lane >> 4;              // 0 or 1 (upper-lane K offset)
        const int l15  = lane & 15;
        v8f c = {};
        #pragma unroll
        for (int kt = 0; kt < 4; ++kt) {         // K = 128, 4 x WMMA k=32
            // A fragment (lane row = l15; elem i -> K = half*16 + i)
            v16h a;
            const _Float16* ap = As + l15 * IN_F + (kt << 5) + (half << 4);
            #pragma unroll
            for (int i = 0; i < 8; ++i) {
                a[i]     = ap[i];
                a[i + 8] = ap[i + 8];
            }
            // B fragment: pre-swizzled, one contiguous 32B vector per lane
            const v16h b = *(const v16h*)(Wf + ((((n0 >> 4) << 2) + kt) << 9) + (lane << 4));
            c = __builtin_amdgcn_wmma_f32_16x16x32_f16(false, a, false, b,
                                                       (short)0, c, false, false);
        }
        // D fragment (ISA: elem v -> row = half*8 + v, col = l15) + bias (+ relu)
        const int col  = n0 + l15;
        const float bv = bias[col];
        #pragma unroll
        for (int v = 0; v < 8; ++v) {
            const int row = m0 + (half << 3) + v;
            float o = c[v] + bv;
            if (do_relu) o = fmaxf(o, 0.0f);
            out[(long)row * out_feats + col] = o;
        }
    }
}

// ----------------------------------------------------------------- launcher
extern "C" void kernel_launch(void* const* d_in, const int* in_sizes, int n_in,
                              void* d_out, int out_size, void* d_ws, size_t ws_size,
                              hipStream_t stream) {
    const float* x   = (const float*)d_in[0];
    const int*   src = (const int*)d_in[1];
    const int*   dst = (const int*)d_in[2];
    const float* W1  = (const float*)d_in[3];
    const float* b1  = (const float*)d_in[4];
    const float* W2  = (const float*)d_in[5];
    const float* b2  = (const float*)d_in[6];
    float* outp = (float*)d_out;

    // workspace layout (256B-aligned slabs)
    char* ws = (char*)d_ws;
    float*    deg = (float*)ws;                                 // 50000 f32 (pad 50176)
    float*    agg = (float*)(ws + 200704);                      // 50000*128 f32
    float*    h1  = (float*)(ws + 200704 + 25600000);           // 50000*128 f32
    _Float16* W1f = (_Float16*)(ws + 200704 + 51200000);        // 16384 f16
    _Float16* W2f = W1f + IN_F * HID_F;                         // 8192 f16

    const int ZB_DEG = (N_NODES + 255) / 256;
    const int ZB_AGG = (N_NODES * IN_F + 255) / 256;
    const int EB     = (N_EDGES + 255) / 256;
    const int SB     = (int)(((long)N_EDGES * 32 + 255) / 256);
    const int MB     = N_NODES / 16;                            // 3125

    // degrees + weight conversion (independent of layer data)
    sage_zero_kernel<<<ZB_DEG, 256, 0, stream>>>(deg, N_NODES);
    sage_deg_kernel<<<EB, 256, 0, stream>>>(dst, deg);
    sage_cvtw_kernel<<<(IN_F * HID_F + 255) / 256, 256, 0, stream>>>(W1, W1f, HID_F);
    sage_cvtw_kernel<<<(IN_F * OUT_F + 255) / 256, 256, 0, stream>>>(W2, W2f, OUT_F);

    // layer 1: agg = scatter(x), h1 = relu(norm(agg,x) @ W1 + b1)
    sage_zero_kernel<<<ZB_AGG, 256, 0, stream>>>(agg, N_NODES * IN_F);
    sage_scatter_kernel<<<SB, 256, 0, stream>>>(x, src, dst, agg);
    sage_gemm_kernel<<<MB, 256, 0, stream>>>(agg, x, deg, W1f, b1, h1, HID_F, 1);

    // layer 2: agg = scatter(h1), out = norm(agg,h1) @ W2 + b2
    sage_zero_kernel<<<ZB_AGG, 256, 0, stream>>>(agg, N_NODES * IN_F);
    sage_scatter_kernel<<<SB, 256, 0, stream>>>(h1, src, dst, agg);
    sage_gemm_kernel<<<MB, 256, 0, stream>>>(agg, h1, deg, W2f, b2, outp, OUT_F, 0);
}
